// TransformerEncoderLayer_19550691132060
// MI455X (gfx1250) — matrix operand order repaired and checked by reading, not compiled, not run on hardware
//
#include <hip/hip_runtime.h>
#include <math.h>

// ---------------------------------------------------------------------------
// Types for CDNA5 WMMA (wave32)
// ---------------------------------------------------------------------------
typedef __attribute__((ext_vector_type(16))) __bf16 v16bf;
typedef __attribute__((ext_vector_type(8)))  __bf16 v8bf;
typedef __attribute__((ext_vector_type(8)))  float  v8f;

// Problem constants (fixed by the reference)
#define BB   4
#define SS   1024
#define DD   1024
#define HH   16
#define HDD  64
#define FF   4096

// ---------------------------------------------------------------------------
// Fragment helpers
// ---------------------------------------------------------------------------
__device__ inline v16bf ld_split(const __bf16* p0, const __bf16* p1) {
    // A-matrix fragment: two 8-element (16B) chunks per lane
    union { v16bf v; v8bf h[2]; } u;
    u.h[0] = *reinterpret_cast<const v8bf*>(p0);
    u.h[1] = *reinterpret_cast<const v8bf*>(p1);
    return u.v;
}
__device__ inline v16bf ld_cont(const __bf16* p) {
    // B-matrix fragment: 16 contiguous elements (32B) per lane
    return *reinterpret_cast<const v16bf*>(p);
}
__device__ inline v8f wmma_bf16(v16bf a, v16bf b, v8f c) {
    return __builtin_amdgcn_wmma_f32_16x16x32_bf16(
        /*neg_a=*/false, a, /*neg_b=*/false, b,
        /*c_mod=*/(short)0, c, /*reuse_a=*/false, /*reuse_b=*/false);
}

// ---------------------------------------------------------------------------
// Kernel 1: transpose + f32 -> bf16 convert for weights.
// src: nmat matrices [R, C] row-major f32 ; dst: nmat matrices [C, R] bf16
// ---------------------------------------------------------------------------
__global__ void k_transpose_cvt(const float* __restrict__ src,
                                __bf16* __restrict__ dst, int R, int C) {
    int mat = blockIdx.y;
    size_t base = (size_t)mat * R * C;
    int total = R * C;
    for (int idx = blockIdx.x * blockDim.x + threadIdx.x; idx < total;
         idx += gridDim.x * blockDim.x) {
        int r = idx / C, c = idx % C;
        dst[base + (size_t)c * R + r] = (__bf16)src[base + idx];
    }
}

// ---------------------------------------------------------------------------
// Kernel 2: LayerNorm over D=1024, one wave (32 lanes) per row, output bf16
// ---------------------------------------------------------------------------
__global__ void k_layernorm_bf16(const float* __restrict__ x,
                                 const float* __restrict__ g,
                                 const float* __restrict__ b,
                                 __bf16* __restrict__ y, int nrows) {
    const int D = DD;
    int wid  = threadIdx.x >> 5;
    int lane = threadIdx.x & 31;
    int row  = blockIdx.x * (blockDim.x >> 5) + wid;
    if (row >= nrows) return;
    const float* xr = x + (size_t)row * D;
    float s = 0.f, s2 = 0.f;
    #pragma unroll
    for (int i = lane; i < D; i += 32) { float v = xr[i]; s += v; s2 += v * v; }
    #pragma unroll
    for (int m = 16; m >= 1; m >>= 1) {
        s  += __shfl_xor(s,  m, 32);
        s2 += __shfl_xor(s2, m, 32);
    }
    float mu  = s / D;
    float var = s2 / D - mu * mu;
    float rs  = rsqrtf(var + 1e-6f);
    __bf16* yr = y + (size_t)row * D;
    #pragma unroll
    for (int i = lane; i < D; i += 32)
        yr[i] = (__bf16)((xr[i] - mu) * rs * g[i] + b[i]);
}

// ---------------------------------------------------------------------------
// Kernel 3: bf16 WMMA GEMM  C[M,N] = A[M,K] * Bt[N,K]^T  (+ epilogue)
//   block = 256 threads = 8 waves (2 x 4), wave tile 32x64 (8 WMMA / K-step),
//   block tile 64x256.  Each B-fragment is reused 2x, each A-fragment 4x.
// MODE 0: + bias, store bf16 scattered to Q/K layout [B,H,S,HD]
// MODE 1: + bias, store bf16 scattered to Vt layout  [B,H,HD,S]
// MODE 2: + bias + residual (f32), store f32 row-major [M,N]
// MODE 3: + bias + exact GELU, store bf16 row-major [M,N]
// ---------------------------------------------------------------------------
template <int MODE>
__global__ void k_gemm_bf16(const __bf16* __restrict__ A,
                            const __bf16* __restrict__ Bt,
                            const float* __restrict__ bias,
                            const float* __restrict__ res,
                            void* __restrict__ outp, int M, int N, int K) {
    int lane = threadIdx.x & 31;
    int warp = threadIdx.x >> 5;
    int wm = warp & 1, wn = warp >> 1;
    int m0 = blockIdx.x * 64  + wm * 32;
    int n0 = blockIdx.y * 256 + wn * 64;
    int half = lane >> 4, lh = lane & 15;

    v8f acc[2][4] = {};

    const __bf16* arow0 = A + (size_t)(m0 + lh) * K;
    const __bf16* arow1 = A + (size_t)(m0 + 16 + lh) * K;
    for (int k0 = 0; k0 < K; k0 += 32) {
        // prefetch A 8 K-steps (256 elements) ahead (global_prefetch_b8)
        if (k0 + 256 < K) __builtin_prefetch(arow0 + k0 + 256, 0, 0);
        v16bf af[2];
        af[0] = ld_split(arow0 + k0 + half * 16, arow0 + k0 + half * 16 + 8);
        af[1] = ld_split(arow1 + k0 + half * 16, arow1 + k0 + half * 16 + 8);
        #pragma unroll
        for (int t = 0; t < 4; ++t) {
            const __bf16* bp =
                Bt + (size_t)(n0 + t * 16 + lh) * K + k0 + half * 16;
            v16bf bf = ld_cont(bp);
            acc[0][t] = wmma_bf16(af[0], bf, acc[0][t]);
            acc[1][t] = wmma_bf16(af[1], bf, acc[1][t]);
        }
    }

    #pragma unroll
    for (int a = 0; a < 2; ++a)
        #pragma unroll
        for (int t = 0; t < 4; ++t) {
            int n = n0 + t * 16 + lh;
            float bv = bias[n];
            #pragma unroll
            for (int r = 0; r < 8; ++r) {
                int m = m0 + a * 16 + half * 8 + r;
                float v = acc[a][t][r] + bv;
                if constexpr (MODE == 0) {          // Q/K: [B,H,S,HD]
                    int bb = m / SS, s = m % SS, hh = n / HDD, hd = n % HDD;
                    ((__bf16*)outp)[((size_t)(bb * HH + hh) * SS + s) * HDD +
                                    hd] = (__bf16)v;
                } else if constexpr (MODE == 1) {   // V transposed: [B,H,HD,S]
                    int bb = m / SS, s = m % SS, hh = n / HDD, hd = n % HDD;
                    ((__bf16*)outp)[((size_t)(bb * HH + hh) * HDD + hd) * SS +
                                    s] = (__bf16)v;
                } else if constexpr (MODE == 2) {   // f32 + residual
                    float* o = (float*)outp;
                    size_t ix = (size_t)m * N + n;
                    o[ix] = v + res[ix];
                } else {                            // bf16 + exact GELU
                    float gl =
                        0.5f * v * (1.0f + erff(v * 0.70710678118654752f));
                    ((__bf16*)outp)[(size_t)m * N + n] = (__bf16)gl;
                }
            }
        }
}

// ---------------------------------------------------------------------------
// Kernel 4: flash attention per (b,h), one wave per 16-query block.
//   Q:[B,H,S,HD] bf16, K:[B,H,S,HD] bf16, Vt:[B,H,HD,S] bf16,
//   mask:[B,S] int, attn out: [B,S,H*HD] bf16 (concat heads)
// ---------------------------------------------------------------------------
__global__ void k_flash_attn(const __bf16* __restrict__ Q,
                             const __bf16* __restrict__ Kb,
                             const __bf16* __restrict__ Vt,
                             const int* __restrict__ mask,
                             __bf16* __restrict__ attn) {
    const float scale = 1.0f / 32.0f;   // 1/sqrt(D), D=1024 per reference
    int lane = threadIdx.x & 31;
    int warp = threadIdx.x >> 5;
    int half = lane >> 4, lh = lane & 15;
    int b = blockIdx.z, h = blockIdx.y;
    int q0 = blockIdx.x * 128 + warp * 16;

    const __bf16* Qp = Q  + (size_t)(b * HH + h) * SS * HDD;
    const __bf16* Kp = Kb + (size_t)(b * HH + h) * SS * HDD;
    const __bf16* Vp = Vt + (size_t)(b * HH + h) * HDD * SS;
    const int* mrow = mask + (size_t)b * SS;

    __shared__ __align__(16) __bf16 Pl[8][16][40];   // per-wave 16x32 (+pad)

    // Q fragments (held across the whole K loop)
    v16bf qa[2];
    const __bf16* qrow = Qp + (size_t)(q0 + lh) * HDD;
    #pragma unroll
    for (int ks = 0; ks < 2; ++ks)
        qa[ks] = ld_split(qrow + ks * 32 + half * 16,
                          qrow + ks * 32 + half * 16 + 8);

    float mi[8], li[8];
    #pragma unroll
    for (int r = 0; r < 8; ++r) { mi[r] = -3.0e38f; li[r] = 0.f; }
    v8f acc[4] = {};

    for (int t0 = 0; t0 < SS; t0 += 32) {
        float sc[2][8];
        #pragma unroll
        for (int tt = 0; tt < 2; ++tt) {
            int tb = t0 + tt * 16;
            v8f c = {};
            #pragma unroll
            for (int ks = 0; ks < 2; ++ks) {
                const __bf16* kp =
                    Kp + (size_t)(tb + lh) * HDD + ks * 32 + half * 16;
                c = wmma_bf16(qa[ks], ld_cont(kp), c);
            }
            float mv = (mrow[tb + lh] == 0) ? -3.0e38f : 0.0f;
            #pragma unroll
            for (int r = 0; r < 8; ++r) sc[tt][r] = c[r] * scale + mv;
        }
        // row max over both tiles, reduce across the 16 lanes of each half
        float rmax[8];
        #pragma unroll
        for (int r = 0; r < 8; ++r) rmax[r] = fmaxf(sc[0][r], sc[1][r]);
        #pragma unroll
        for (int m = 1; m < 16; m <<= 1)
            #pragma unroll
            for (int r = 0; r < 8; ++r)
                rmax[r] = fmaxf(rmax[r], __shfl_xor(rmax[r], m, 32));

        float corr[8], rsum[8];
        #pragma unroll
        for (int r = 0; r < 8; ++r) {
            float mn = fmaxf(mi[r], rmax[r]);
            corr[r] = __expf(mi[r] - mn);
            mi[r] = mn;
            rsum[r] = 0.f;
        }
        // p = exp(s - m), write into per-wave LDS tile (C-layout -> A-layout)
        #pragma unroll
        for (int tt = 0; tt < 2; ++tt)
            #pragma unroll
            for (int r = 0; r < 8; ++r) {
                float p = __expf(sc[tt][r] - mi[r]);
                rsum[r] += p;
                Pl[warp][half * 8 + r][tt * 16 + lh] = (__bf16)p;
            }
        #pragma unroll
        for (int m = 1; m < 16; m <<= 1)
            #pragma unroll
            for (int r = 0; r < 8; ++r)
                rsum[r] += __shfl_xor(rsum[r], m, 32);
        #pragma unroll
        for (int r = 0; r < 8; ++r) li[r] = li[r] * corr[r] + rsum[r];
        #pragma unroll
        for (int t = 0; t < 4; ++t)
            #pragma unroll
            for (int r = 0; r < 8; ++r) acc[t][r] *= corr[r];

        __syncthreads();   // ensure LDS stores visible before A-fragment read

        v16bf pa = ld_split(&Pl[warp][lh][half * 16],
                            &Pl[warp][lh][half * 16 + 8]);
        #pragma unroll
        for (int n = 0; n < 4; ++n) {
            const __bf16* vp =
                Vp + (size_t)(n * 16 + lh) * SS + t0 + half * 16;
            acc[n] = wmma_bf16(pa, ld_cont(vp), acc[n]);
        }
        __syncthreads();
    }

    // write concat-head output [B, S, H*HD] bf16
    #pragma unroll
    for (int n = 0; n < 4; ++n)
        #pragma unroll
        for (int r = 0; r < 8; ++r) {
            int s  = q0 + half * 8 + r;
            int hd = n * 16 + lh;
            float v = acc[n][r] / li[r];
            attn[((size_t)(b * SS + s)) * (HH * HDD) + h * HDD + hd] =
                (__bf16)v;
        }
}

// ---------------------------------------------------------------------------
// Host-side orchestration
// ---------------------------------------------------------------------------
extern "C" void kernel_launch(void* const* d_in, const int* in_sizes, int n_in,
                              void* d_out, int out_size, void* d_ws,
                              size_t ws_size, hipStream_t stream) {
    const float* x    = (const float*)d_in[0];
    const int*   mask = (const int*)  d_in[1];
    const float* wq   = (const float*)d_in[2];
    const float* bq   = (const float*)d_in[3];
    const float* wk   = (const float*)d_in[4];
    const float* bk   = (const float*)d_in[5];
    const float* wv   = (const float*)d_in[6];
    const float* bv   = (const float*)d_in[7];
    const float* wo   = (const float*)d_in[8];
    const float* bo   = (const float*)d_in[9];
    const float* g1   = (const float*)d_in[10];
    const float* be1  = (const float*)d_in[11];
    const float* g2   = (const float*)d_in[12];
    const float* be2  = (const float*)d_in[13];
    const float* w1   = (const float*)d_in[14];
    const float* b1   = (const float*)d_in[15];
    const float* w2   = (const float*)d_in[16];
    const float* b2   = (const float*)d_in[17];

    const int M = BB * SS;   // 4096 rows

    // bump allocator on workspace
    char* wsp = (char*)d_ws;
    size_t off = 0;
    auto alloc = [&](size_t bytes) -> void* {
        void* p = wsp + off;
        off += (bytes + 255) & ~(size_t)255;
        return p;
    };
    __bf16* wqT  = (__bf16*)alloc((size_t)HH * DD * HDD * 2);
    __bf16* wkT  = (__bf16*)alloc((size_t)HH * DD * HDD * 2);
    __bf16* wvT  = (__bf16*)alloc((size_t)HH * DD * HDD * 2);
    __bf16* woT  = (__bf16*)alloc((size_t)DD * HH * HDD * 2);
    __bf16* w1T  = (__bf16*)alloc((size_t)FF * DD * 2);
    __bf16* w2T  = (__bf16*)alloc((size_t)DD * FF * 2);
    __bf16* y1   = (__bf16*)alloc((size_t)M * DD * 2);
    __bf16* Qb   = (__bf16*)alloc((size_t)M * HH * HDD * 2 / BB * BB);
    __bf16* Kbuf = (__bf16*)alloc((size_t)M * HH * HDD * 2 / BB * BB);
    __bf16* Vtb  = (__bf16*)alloc((size_t)M * HH * HDD * 2 / BB * BB);
    __bf16* attn = (__bf16*)alloc((size_t)M * HH * HDD * 2);
    float*  x1   = (float*) alloc((size_t)M * DD * 4);
    __bf16* y2   = (__bf16*)alloc((size_t)M * DD * 2);
    __bf16* h1   = (__bf16*)alloc((size_t)M * FF * 2);

    // --- 1) weight transpose+convert ---
    {
        dim3 g((DD * HDD + 255) / 256, HH);
        k_transpose_cvt<<<g, 256, 0, stream>>>(wq, wqT, DD, HDD);
        k_transpose_cvt<<<g, 256, 0, stream>>>(wk, wkT, DD, HDD);
        k_transpose_cvt<<<g, 256, 0, stream>>>(wv, wvT, DD, HDD);
    }
    {
        dim3 g((HH * HDD * DD + 255) / 256, 1);
        k_transpose_cvt<<<g, 256, 0, stream>>>(wo, woT, HH * HDD, DD);
    }
    {
        dim3 g((DD * FF + 255) / 256, 1);
        k_transpose_cvt<<<g, 256, 0, stream>>>(w1, w1T, DD, FF);
        k_transpose_cvt<<<g, 256, 0, stream>>>(w2, w2T, FF, DD);
    }

    // --- 2) LN1 -> y1 (bf16) ---
    k_layernorm_bf16<<<M / 8, 256, 0, stream>>>(x, g1, be1, y1, M);

    // --- 3) QKV projections (WMMA GEMMs, fused bias + layout scatter) ---
    {
        dim3 g(M / 64, (HH * HDD) / 256);
        k_gemm_bf16<0><<<g, 256, 0, stream>>>(y1, wqT, bq, nullptr, Qb,
                                              M, HH * HDD, DD);
        k_gemm_bf16<0><<<g, 256, 0, stream>>>(y1, wkT, bk, nullptr, Kbuf,
                                              M, HH * HDD, DD);
        k_gemm_bf16<1><<<g, 256, 0, stream>>>(y1, wvT, bv, nullptr, Vtb,
                                              M, HH * HDD, DD);
    }

    // --- 4) flash attention ---
    {
        dim3 g(SS / 128, HH, BB);
        k_flash_attn<<<g, 256, 0, stream>>>(Qb, Kbuf, Vtb, mask, attn);
    }

    // --- 5) output projection + residual -> x1 (f32) ---
    {
        dim3 g(M / 64, DD / 256);
        k_gemm_bf16<2><<<g, 256, 0, stream>>>(attn, woT, bo, x, x1,
                                              M, DD, HH * HDD);
    }

    // --- 6) LN2 -> y2 (bf16) ---
    k_layernorm_bf16<<<M / 8, 256, 0, stream>>>(x1, g2, be2, y2, M);

    // --- 7) FFN1 + exact GELU -> h1 (bf16) ---
    {
        dim3 g(M / 64, FF / 256);
        k_gemm_bf16<3><<<g, 256, 0, stream>>>(y2, w1T, b1, nullptr, h1,
                                              M, FF, DD);
    }

    // --- 8) FFN2 + residual -> d_out (f32) ---
    {
        dim3 g(M / 64, DD / 256);
        k_gemm_bf16<2><<<g, 256, 0, stream>>>(h1, w2T, b2, x1, (float*)d_out,
                                              M, DD, FF);
    }
}
